// NetGCN_15324443312398
// MI455X (gfx1250) — matrix operand order repaired and checked by reading, not compiled, not run on hardware
//
#include <hip/hip_runtime.h>
#include <hip/hip_bf16.h>
#include <stdint.h>

// CDNA5 wave32 WMMA fragment types
typedef __attribute__((ext_vector_type(16))) __bf16 v16bf;
typedef __attribute__((ext_vector_type(8)))  float  v8f;
typedef __attribute__((ext_vector_type(4)))  int    v4i;

union Frag16 { uint4 u[2]; v16bf v; };  // 32 bytes: 16 bf16 elements

#if __has_builtin(__builtin_amdgcn_global_load_async_to_lds_b128)
#define HAVE_ASYNC_LDS 1
#endif

// Address-space helpers (AS1: same 64-bit address; AS3: low 32 bits = LDS offset)
__device__ inline __attribute__((address_space(1))) v4i* as_global_v4i(const void* p) {
    return (__attribute__((address_space(1))) v4i*)(uintptr_t)p;
}
__device__ inline __attribute__((address_space(3))) v4i* as_lds_v4i(void* p) {
    return (__attribute__((address_space(3))) v4i*)(uint32_t)(uintptr_t)p;
}

// ---------------------------------------------------------------------------
// utility kernels
// ---------------------------------------------------------------------------
__global__ void k_zero_f32(float* __restrict__ p, long n) {
    long i = (long)blockIdx.x * blockDim.x + threadIdx.x;
    if (i < n) p[i] = 0.0f;
}

__global__ void k_degree(const int* __restrict__ si, const int* __restrict__ di,
                         const int* __restrict__ sb, const int* __restrict__ db,
                         float* __restrict__ dsi, float* __restrict__ ddi,
                         float* __restrict__ dsb, float* __restrict__ ddb, int E) {
    int t = blockIdx.x * blockDim.x + threadIdx.x;
    if (t < E) {
        atomicAdd(dsi + si[t], 1.0f);
        atomicAdd(ddi + di[t], 1.0f);
        atomicAdd(dsb + sb[t], 1.0f);
        atomicAdd(ddb + db[t], 1.0f);
    }
}

__global__ void k_finish_deg(float* __restrict__ d, int n) {
    int i = blockIdx.x * blockDim.x + threadIdx.x;
    if (i < n) d[i] = rsqrtf(fmaxf(d[i], 1.0f));   // deg^{-1/2}, clip(deg,1)
}

// Wt[n*128+k] = bf16(W[k*128+n])  (pre-transpose + convert, 128x128)
__global__ void k_prep_w(const float* __restrict__ W, __hip_bfloat16* __restrict__ Wt) {
    int i = blockIdx.x * blockDim.x + threadIdx.x;
    if (i < 128 * 128) {
        int k = i >> 7, n = i & 127;
        Wt[n * 128 + k] = __float2bfloat16(W[i]);
    }
}

// out[row,f] = bf16( x[row,f] * dinv[row] ), zero for pad rows
__global__ void k_scale_cast(const float* __restrict__ x, const float* __restrict__ dinv,
                             __hip_bfloat16* __restrict__ out, int Nrows, long total) {
    long i = (long)blockIdx.x * blockDim.x + threadIdx.x;
    if (i >= total) return;
    int row = (int)(i >> 7);
    float v = 0.0f;
    if (row < Nrows) v = x[i] * dinv[row];
    out[i] = __float2bfloat16(v);
}

// ---------------------------------------------------------------------------
// GEMM: C[Np x 128] = A_bf16[Np x 128] @ W[128 x 128]  (fp32 accumulate)
// Wt is pre-transposed bf16 [n][k]. 256 threads = 8 waves; each wave owns a
// 32-row x 128-col strip (two A fragments, 16 accumulator tiles), so every
// LDS B-fragment load feeds TWO back-to-back v_wmma ops.
// Wt is staged to LDS with GLOBAL_LOAD_ASYNC_TO_LDS_B128 when available.
// ---------------------------------------------------------------------------
__global__ __launch_bounds__(256) void k_gemm_wmma(
        const __hip_bfloat16* __restrict__ A,    // [Np x 128], row-major
        const __hip_bfloat16* __restrict__ Wt,   // [128 x 128], n-major (pre-transposed)
        float*                __restrict__ C,    // [Np x 128]
        int Np) {
    __shared__ __align__(32) __hip_bfloat16 WtL[128 * 128];  // 32 KB

    const int tid = threadIdx.x;
#if defined(HAVE_ASYNC_LDS) && defined(__AMDGCN__)
    // async stage: 32 KB = 2048 x b128; 256 lanes x 8 iterations
    {
        const char* g = (const char*)Wt;
        char* l = (char*)WtL;
#pragma unroll
        for (int it = 0; it < 8; ++it) {
            int off = (it * 256 + tid) * 16;
            __builtin_amdgcn_global_load_async_to_lds_b128(
                as_global_v4i(g + off), as_lds_v4i(l + off), 0, 0);
        }
#if __has_builtin(__builtin_amdgcn_s_wait_asynccnt)
        __builtin_amdgcn_s_wait_asynccnt(0);
#else
        asm volatile("s_wait_asynccnt 0x0" ::: "memory");
#endif
    }
#else
    {
        const uint4* g = (const uint4*)Wt;   // 2048 x 16B
        uint4* l = (uint4*)WtL;
        for (int i = tid; i < 2048; i += 256) l[i] = g[i];
    }
#endif
    __syncthreads();

    const int wave = tid >> 5;
    const int lane = tid & 31;
    const int lo = lane & 15;          // row-in-tile (A) / col-in-tile (B,C)
    const int hi = lane >> 4;          // half-select per ISA layouts
    const int m0 = blockIdx.x * 256 + wave * 32;
    if (m0 >= Np) return;              // Np multiple of 256; uniform branch

    v8f acc0[8] = {};                  // rows m0   .. m0+15
    v8f acc1[8] = {};                  // rows m0+16.. m0+31

    const __hip_bfloat16* arow0 = A + (long)(m0 + lo) * 128;
    const __hip_bfloat16* arow1 = arow0 + 16 * 128;

#pragma unroll
    for (int kt = 0; kt < 4; kt++) {   // K = 4 x 32
        // A fragment 16x32: lane<16 holds K {0..7,16..23}, lane>=16 holds {8..15,24..31}
        const int ka = kt * 32 + hi * 8;
        Frag16 a0, a1;
        a0.u[0] = *(const uint4*)(arow0 + ka);
        a0.u[1] = *(const uint4*)(arow0 + ka + 16);
        a1.u[0] = *(const uint4*)(arow1 + ka);
        a1.u[1] = *(const uint4*)(arow1 + ka + 16);
#pragma unroll
        for (int nt = 0; nt < 8; nt++) {
            // B fragment 32x16: lane n = nt*16+lo, lane<16 holds K {0..7,16..23}, lane>=16 holds {8..15,24..31}
            const __hip_bfloat16* bp = WtL + (long)(nt * 16 + lo) * 128 + kt * 32 + hi * 8;
            Frag16 bf;
            bf.u[0] = *(const uint4*)bp;
            bf.u[1] = *(const uint4*)(bp + 16);
            acc0[nt] = __builtin_amdgcn_wmma_f32_16x16x32_bf16(
                false, a0.v, false, bf.v, (short)0, acc0[nt], false, false);
            acc1[nt] = __builtin_amdgcn_wmma_f32_16x16x32_bf16(
                false, a1.v, false, bf.v, (short)0, acc1[nt], false, false);
        }
    }

    // C/D layout: VGPR r -> row base + hi*8 + r, col nt*16 + lo
#pragma unroll
    for (int nt = 0; nt < 8; nt++) {
        float* c0 = C + (long)(m0 + hi * 8) * 128 + nt * 16 + lo;
        float* c1 = c0 + 16 * 128;
#pragma unroll
        for (int r = 0; r < 8; r++) {
            c0[(long)r * 128] = acc0[nt][r];
            c1[(long)r * 128] = acc1[nt][r];
        }
    }
}

// ---------------------------------------------------------------------------
// SpMM scatter-add: one wave per edge, float4 per lane (128 feats / 32 lanes)
// ---------------------------------------------------------------------------
__global__ __launch_bounds__(256) void k_spmm(
        const float* __restrict__ h, const int* __restrict__ src,
        const int* __restrict__ dst, float* __restrict__ agg, int E) {
    int w = blockIdx.x * 8 + (threadIdx.x >> 5);
    if (w >= E) return;
    int lane = threadIdx.x & 31;
    long s = src[w];
    long d = dst[w];
    const float4 hv = *(const float4*)(h + s * 128 + lane * 4);
    float* ap = agg + d * 128 + lane * 4;
    atomicAdd(ap + 0, hv.x);
    atomicAdd(ap + 1, hv.y);
    atomicAdd(ap + 2, hv.z);
    atomicAdd(ap + 3, hv.w);
}

// h1 = relu(agg1*ddi + b1i) + relu(agg2*ddb + b1b);  hs = bf16(h1 * dsi);
// also clears agg1 (reused for layer-2 aggregation). Pad rows -> 0.
__global__ void k_combine(const float* __restrict__ agg1, const float* __restrict__ agg2,
                          const float* __restrict__ ddi, const float* __restrict__ ddb,
                          const float* __restrict__ dsi,
                          const float* __restrict__ b1i, const float* __restrict__ b1b,
                          __hip_bfloat16* __restrict__ hs, float* __restrict__ clr,
                          int Nrows, long total) {
    long i = (long)blockIdx.x * blockDim.x + threadIdx.x;
    if (i >= total) return;
    int row = (int)(i >> 7), f = (int)(i & 127);
    float v = 0.0f;
    if (row < Nrows) {
        float y1 = fmaxf(fmaf(agg1[i], ddi[row], b1i[f]), 0.0f);
        float y2 = fmaxf(fmaf(agg2[i], ddb[row], b1b[f]), 0.0f);
        v = (y1 + y2) * dsi[row];
        clr[i] = 0.0f;                 // re-zero agg1 for the layer-2 SpMM
    }
    hs[i] = __float2bfloat16(v);
}

__global__ void k_final(const float* __restrict__ agg, const float* __restrict__ ddi,
                        const float* __restrict__ b2, float* __restrict__ out, long total) {
    long i = (long)blockIdx.x * blockDim.x + threadIdx.x;
    if (i >= total) return;
    int row = (int)(i >> 7), f = (int)(i & 127);
    out[i] = fmaf(agg[i], ddi[row], b2[f]);
}

// ---------------------------------------------------------------------------
extern "C" void kernel_launch(void* const* d_in, const int* in_sizes, int n_in,
                              void* d_out, int out_size, void* d_ws, size_t ws_size,
                              hipStream_t stream) {
    const float* x     = (const float*)d_in[0];
    const int*   src_i = (const int*)  d_in[1];
    const int*   dst_i = (const int*)  d_in[2];
    const int*   src_b = (const int*)  d_in[3];
    const int*   dst_b = (const int*)  d_in[4];
    const float* W1i   = (const float*)d_in[5];
    const float* b1i   = (const float*)d_in[6];
    const float* W1b   = (const float*)d_in[7];
    const float* b1b   = (const float*)d_in[8];
    const float* W2    = (const float*)d_in[9];
    const float* b2    = (const float*)d_in[10];
    float* out = (float*)d_out;

    const int  N   = in_sizes[0] / 128;
    const int  E   = in_sizes[1];
    const int  Np  = ((N + 255) / 256) * 256;     // pad rows: 256 per GEMM block
    const long NPF = (long)Np * 128;
    const long NF  = (long)N * 128;

    // workspace layout (ws base is 256B aligned)
    char* w = (char*)d_ws;
    float* dsi = (float*)w;  // deg^{-1/2}: src_i | dst_i | src_b | dst_b
    float* ddi = dsi + N;
    float* dsb = dsi + 2 * (long)N;
    float* ddb = dsi + 3 * (long)N;
    w += sizeof(float) * 4 * (long)N;
    w = (char*)(((uintptr_t)w + 255) & ~(uintptr_t)255);
    __hip_bfloat16* Wt1i = (__hip_bfloat16*)w;  w += sizeof(__hip_bfloat16) * 128 * 128;
    __hip_bfloat16* Wt1b = (__hip_bfloat16*)w;  w += sizeof(__hip_bfloat16) * 128 * 128;
    __hip_bfloat16* Wt2  = (__hip_bfloat16*)w;  w += sizeof(__hip_bfloat16) * 128 * 128;
    __hip_bfloat16* Abf  = (__hip_bfloat16*)w;  w += sizeof(__hip_bfloat16) * NPF;
    float* H1  = (float*)w;                     w += sizeof(float) * NPF;
    float* AG1 = (float*)w;                     w += sizeof(float) * NF;
    float* AG2 = (float*)w;                     /* w += sizeof(float) * NF; */

    const int B = 256;
    // 1) zero degree counters + both aggregation buffers (AG1,AG2 contiguous)
    k_zero_f32<<<(int)((4L * N + B - 1) / B), B, 0, stream>>>(dsi, 4L * N);
    k_zero_f32<<<(int)((2 * NF + B - 1) / B), B, 0, stream>>>(AG1, 2 * NF);
    // 2) pre-transpose + convert the three weight matrices to bf16
    k_prep_w<<<64, B, 0, stream>>>(W1i, Wt1i);
    k_prep_w<<<64, B, 0, stream>>>(W1b, Wt1b);
    k_prep_w<<<64, B, 0, stream>>>(W2,  Wt2);
    // 3) degrees -> deg^{-1/2}
    k_degree<<<(E + B - 1) / B, B, 0, stream>>>(src_i, dst_i, src_b, dst_b,
                                                dsi, ddi, dsb, ddb, E);
    k_finish_deg<<<(4 * N + B - 1) / B, B, 0, stream>>>(dsi, 4 * N);

    // ---- conv1, relation 'interacts' ----
    k_scale_cast<<<(int)((NPF + B - 1) / B), B, 0, stream>>>(x, dsi, Abf, N, NPF);
    k_gemm_wmma<<<Np / 256, B, 0, stream>>>(Abf, Wt1i, H1, Np);
    k_spmm<<<(E + 7) / 8, B, 0, stream>>>(H1, src_i, dst_i, AG1, E);

    // ---- conv1, relation 'behave' (reuse Abf / H1) ----
    k_scale_cast<<<(int)((NPF + B - 1) / B), B, 0, stream>>>(x, dsb, Abf, N, NPF);
    k_gemm_wmma<<<Np / 256, B, 0, stream>>>(Abf, Wt1b, H1, Np);
    k_spmm<<<(E + 7) / 8, B, 0, stream>>>(H1, src_b, dst_b, AG2, E);

    // ---- combine + ReLU + rescale for conv2; clears AG1 ----
    k_combine<<<(int)((NPF + B - 1) / B), B, 0, stream>>>(
        AG1, AG2, ddi, ddb, dsi, b1i, b1b, Abf, AG1, N, NPF);

    // ---- conv2 ('interacts' only, no activation) ----
    k_gemm_wmma<<<Np / 256, B, 0, stream>>>(Abf, Wt2, H1, Np);
    k_spmm<<<(E + 7) / 8, B, 0, stream>>>(H1, src_i, dst_i, AG1, E);
    k_final<<<(int)((NF + B - 1) / B), B, 0, stream>>>(AG1, ddi, b2, out, NF);
}
